// APPNP_19739669692445
// MI455X (gfx1250) — matrix operand order repaired and checked by reading, not compiled, not run on hardware
//
#include <hip/hip_runtime.h>

typedef __attribute__((ext_vector_type(16))) __bf16 v16bf;
typedef __attribute__((ext_vector_type(8)))  float  v8f;

#define IN_F   512
#define HID_F  256
#define OUT_F  64
#define ALPHA  0.01f
#define KITER  10
#define BROW   520   // padded LDS row stride in bf16 units (1040 B)

__device__ __forceinline__ unsigned short f2bf(float f) {
    union { float f; unsigned u; } c; c.f = f;
    unsigned u = c.u;
    unsigned r = u + 0x7FFFu + ((u >> 16) & 1u);   // round-to-nearest-even
    return (unsigned short)(r >> 16);
}

union Frag16 {
    v16bf v;
    uint4 q[2];
    unsigned short u[16];
};

// ---------------- x fp32 -> bf16 ----------------
__global__ void k_cvtx(const float* __restrict__ x, unsigned short* __restrict__ xb) {
    size_t i = ((size_t)blockIdx.x * blockDim.x + threadIdx.x) * 8;
    float4 f0 = *(const float4*)(x + i);
    float4 f1 = *(const float4*)(x + i + 4);
    uint4 o;
    o.x = f2bf(f0.x) | ((unsigned)f2bf(f0.y) << 16);
    o.y = f2bf(f0.z) | ((unsigned)f2bf(f0.w) << 16);
    o.z = f2bf(f1.x) | ((unsigned)f2bf(f1.y) << 16);
    o.w = f2bf(f1.z) | ((unsigned)f2bf(f1.w) << 16);
    *(uint4*)(xb + i) = o;
}

// ---------------- weight prep: transpose + fp32->bf16 ----------------
__global__ void k_w1t(const float* __restrict__ W1, unsigned short* __restrict__ w1t) {
    int i = blockIdx.x * blockDim.x + threadIdx.x;   // i = k*256 + n
    int k = i >> 8;
    int n = i & 255;
    w1t[(size_t)n * IN_F + k] = f2bf(W1[i]);
}

__global__ void k_w2t(const float* __restrict__ W2, unsigned short* __restrict__ w2t) {
    int i = blockIdx.x * blockDim.x + threadIdx.x;   // i = k*64 + n
    int k = i >> 6;
    int n = i & 63;
    w2t[(size_t)n * HID_F + k] = f2bf(W2[i]);
}

// ---------------- GEMM1: h1 = relu(x @ W1 + b1), bf16 out ----------------
__global__ void __launch_bounds__(256) k_gemm1(const unsigned short* __restrict__ xb,
                                               const unsigned short* __restrict__ w1t,
                                               const float* __restrict__ b1,
                                               unsigned short* __restrict__ h1,
                                               int MT) {
    extern __shared__ unsigned short smB[];           // 128 * BROW bf16 = 133120 B
    const int tid = threadIdx.x;
    const int n0 = blockIdx.y << 7;                   // column group base (0 / 128)

    // cooperative stage: 128 rows x 512 bf16 (8192 x 16B chunks, 32/thread)
    #pragma unroll
    for (int i = 0; i < 32; ++i) {
        int idx = i * 256 + tid;
        int rown = idx >> 6;
        int ch   = idx & 63;
        *(uint4*)(&smB[rown * BROW + ch * 8]) =
            *(const uint4*)(w1t + (size_t)(n0 + rown) * IN_F + ch * 8);
    }
    __syncthreads();

    const int wave = tid >> 5;
    const int lane = tid & 31;
    const int mt = blockIdx.x * 8 + wave;
    if (mt >= MT) return;                             // wave-uniform, after barrier
    const int m0 = mt << 4;
    const int half = lane >> 4;
    const int lr = lane & 15;

    v8f c[8] = {};
    const unsigned short* arow = xb + (size_t)(m0 + lr) * IN_F;

    for (int ks = 0; ks < 16; ++ks) {
        Frag16 a;
        const unsigned short* p0 = arow + ks * 32 + half * 8;
        a.q[0] = *(const uint4*)(p0);
        a.q[1] = *(const uint4*)(p0 + 16);

        Frag16 b[8];
        const unsigned short* bbase = &smB[ks * 32 + half * 8];
        #pragma unroll
        for (int t = 0; t < 8; ++t) {
            const unsigned short* bp = bbase + (t * 16 + lr) * BROW;
            b[t].q[0] = *(const uint4*)(bp);
            b[t].q[1] = *(const uint4*)(bp + 16);
        }
        #pragma unroll
        for (int t = 0; t < 8; ++t)
            c[t] = __builtin_amdgcn_wmma_f32_16x16x32_bf16(
                false, a.v, false, b[t].v, (short)0, c[t], false, false);
    }

    #pragma unroll
    for (int t = 0; t < 8; ++t) {
        int gn = n0 + t * 16 + lr;
        float bias = b1[gn];
        #pragma unroll
        for (int r = 0; r < 8; ++r) {
            int gm = m0 + r + half * 8;
            float v = c[t][r] + bias;
            v = v > 0.f ? v : 0.f;
            h1[(size_t)gm * HID_F + gn] = f2bf(v);
        }
    }
}

// ---------------- GEMM2: h0 = h1 @ W2 + b2, fp32 out ----------------
__global__ void k_gemm2(const unsigned short* __restrict__ h1,
                        const unsigned short* __restrict__ w2t,
                        const float* __restrict__ b2,
                        float* __restrict__ out) {
    const int lane = threadIdx.x & 31;
    const int wavew = blockIdx.x * 2 + (threadIdx.x >> 5);
    const int m0 = wavew << 4;
    const int half = lane >> 4;
    const int lr = lane & 15;

    v8f c[4] = {};
    const unsigned short* arow = h1 + (size_t)(m0 + lr) * HID_F;

    for (int ks = 0; ks < 8; ++ks) {
        Frag16 a;
        const unsigned short* p0 = arow + ks * 32 + half * 8;
        a.q[0] = *(const uint4*)(p0);
        a.q[1] = *(const uint4*)(p0 + 16);

        Frag16 b[4];
        #pragma unroll
        for (int t = 0; t < 4; ++t) {
            const unsigned short* bp =
                w2t + (size_t)(t * 16 + lr) * HID_F + ks * 32 + half * 8;
            b[t].q[0] = *(const uint4*)(bp);
            b[t].q[1] = *(const uint4*)(bp + 16);
        }
        #pragma unroll
        for (int t = 0; t < 4; ++t)
            c[t] = __builtin_amdgcn_wmma_f32_16x16x32_bf16(
                false, a.v, false, b[t].v, (short)0, c[t], false, false);
    }
    #pragma unroll
    for (int t = 0; t < 4; ++t) {
        int gn = t * 16 + lr;
        float bias = b2[gn];
        #pragma unroll
        for (int r = 0; r < 8; ++r) {
            int gm = m0 + r + half * 8;
            out[(size_t)gm * OUT_F + gn] = c[t][r] + bias;
        }
    }
}

// ---------------- CSR build ----------------
__global__ void k_zero(int* __restrict__ p, int n) {
    int i = blockIdx.x * blockDim.x + threadIdx.x;
    if (i < n) p[i] = 0;
}

__global__ void k_count(const int* __restrict__ row, int* __restrict__ cnt, int E_) {
    int e = blockIdx.x * blockDim.x + threadIdx.x;
    if (e < E_) atomicAdd(&cnt[row[e]], 1);
}

__global__ void k_scan1(const int* __restrict__ cnt, int* __restrict__ rp,
                        int* __restrict__ part, int n) {
    __shared__ int sm[1024];
    int t = threadIdx.x;
    int idx = blockIdx.x * 1024 + t;
    int v = (idx < n) ? cnt[idx] : 0;
    sm[t] = v;
    __syncthreads();
    for (int off = 1; off < 1024; off <<= 1) {
        int add = (t >= off) ? sm[t - off] : 0;
        __syncthreads();
        sm[t] += add;
        __syncthreads();
    }
    if (idx < n) rp[idx] = sm[t] - v;
    if (t == 1023) part[blockIdx.x] = sm[t];
}

__global__ void k_scan2(const int* __restrict__ part, int* __restrict__ pexcl, int nb) {
    __shared__ int sm[128];
    int t = threadIdx.x;
    int v = (t < nb) ? part[t] : 0;
    sm[t] = v;
    __syncthreads();
    for (int off = 1; off < 128; off <<= 1) {
        int add = (t >= off) ? sm[t - off] : 0;
        __syncthreads();
        sm[t] += add;
        __syncthreads();
    }
    pexcl[t] = sm[t] - v;
}

__global__ void k_scan3(int* __restrict__ rp, const int* __restrict__ pexcl,
                        int* __restrict__ wp, int n, int E_) {
    int idx = blockIdx.x * blockDim.x + threadIdx.x;
    if (idx < n) {
        int v = rp[idx] + pexcl[idx >> 10];
        rp[idx] = v;
        wp[idx] = v;
    }
    if (idx == 0) rp[n] = E_;
}

// packs (col, weight_bits) into one int2 per edge -> single b64 load in prop
__global__ void k_scatter(const int* __restrict__ row, const int* __restrict__ col,
                          const float* __restrict__ ew, int* __restrict__ wp,
                          int2* __restrict__ ed, int E_) {
    int e = blockIdx.x * blockDim.x + threadIdx.x;
    if (e >= E_) return;
    int r = row[e];
    int pos = atomicAdd(&wp[r], 1);
    int2 pk;
    pk.x = col[e];
    pk.y = __float_as_int(ew[e]);
    ed[pos] = pk;
}

// ---------------- propagation: out[r] = (1-a)*sum_e w_e*h[col_e] + a*h[r] -------
// one wave32 per row; lane owns features {2*lane, 2*lane+1}; 2 edges in flight
__global__ void k_prop(const float* __restrict__ hin, float* __restrict__ hout,
                       const int* __restrict__ rp, const int2* __restrict__ ed) {
    int wave = (blockIdx.x * blockDim.x + threadIdx.x) >> 5;
    int lane = threadIdx.x & 31;
    int r = wave;
    int f = lane << 1;
    int s = rp[r], e = rp[r + 1];
    float2 hs = *(const float2*)(hin + (size_t)r * OUT_F + f);
    float ax = 0.f, ay = 0.f;
    int j = s;
    for (; j + 1 < e; j += 2) {
        int2 ea = ed[j];
        int2 eb = ed[j + 1];
        float2 ha = *(const float2*)(hin + (size_t)ea.x * OUT_F + f);
        float2 hb = *(const float2*)(hin + (size_t)eb.x * OUT_F + f);
        float wa = __int_as_float(ea.y);
        float wb = __int_as_float(eb.y);
        ax = fmaf(wa, ha.x, ax);
        ay = fmaf(wa, ha.y, ay);
        ax = fmaf(wb, hb.x, ax);
        ay = fmaf(wb, hb.y, ay);
    }
    if (j < e) {
        int2 ea = ed[j];
        float2 ha = *(const float2*)(hin + (size_t)ea.x * OUT_F + f);
        float wa = __int_as_float(ea.y);
        ax = fmaf(wa, ha.x, ax);
        ay = fmaf(wa, ha.y, ay);
    }
    float2 o;
    o.x = fmaf(1.f - ALPHA, ax, ALPHA * hs.x);
    o.y = fmaf(1.f - ALPHA, ay, ALPHA * hs.y);
    *(float2*)(hout + (size_t)r * OUT_F + f) = o;
}

// ---------------- launch ----------------
extern "C" void kernel_launch(void* const* d_in, const int* in_sizes, int n_in,
                              void* d_out, int out_size, void* d_ws, size_t ws_size,
                              hipStream_t stream) {
    const float* x  = (const float*)d_in[0];
    const float* W1 = (const float*)d_in[1];
    const float* b1 = (const float*)d_in[2];
    const float* W2 = (const float*)d_in[3];
    const float* b2 = (const float*)d_in[4];
    const float* ew = (const float*)d_in[5];
    const int*  row = (const int*)d_in[6];
    const int*  col = (const int*)d_in[7];
    float* out = (float*)d_out;

    const int N  = in_sizes[0] / IN_F;   // 100000
    const int E  = in_sizes[5];          // 3200000
    const int MT = N / 16;               // 6250

    char* p = (char*)d_ws;
    auto carve = [&](size_t bytes) {
        void* q = (void*)p;
        p += (bytes + 255) & ~(size_t)255;
        return q;
    };
    unsigned short* xb  = (unsigned short*)carve((size_t)N * IN_F * 2);
    unsigned short* w1t = (unsigned short*)carve((size_t)HID_F * IN_F * 2);
    unsigned short* w2t = (unsigned short*)carve((size_t)OUT_F * HID_F * 2);
    unsigned short* h1  = (unsigned short*)carve((size_t)N * HID_F * 2);
    float* hA  = (float*)carve((size_t)N * OUT_F * 4);
    float* hB  = (float*)carve((size_t)N * OUT_F * 4);
    int* cnt   = (int*)carve((size_t)N * 4);
    int* rp    = (int*)carve((size_t)(N + 1) * 4);
    int* wp    = (int*)carve((size_t)N * 4);
    int* part  = (int*)carve(128 * 4);
    int* pexcl = (int*)carve(128 * 4);
    int2* ed   = (int2*)carve((size_t)E * 8);

    // input prep
    k_cvtx<<<(int)(((size_t)N * IN_F / 8) / 256), 256, 0, stream>>>(x, xb);
    k_w1t<<<(IN_F * HID_F) / 256, 256, 0, stream>>>(W1, w1t);
    k_w2t<<<(HID_F * OUT_F) / 256, 256, 0, stream>>>(W2, w2t);

    // CSR build
    int nb = (N + 1023) / 1024;
    k_zero<<<(N + 255) / 256, 256, 0, stream>>>(cnt, N);
    k_count<<<(E + 255) / 256, 256, 0, stream>>>(row, cnt, E);
    k_scan1<<<nb, 1024, 0, stream>>>(cnt, rp, part, N);
    k_scan2<<<1, 128, 0, stream>>>(part, pexcl, nb);
    k_scan3<<<(N + 255) / 256, 256, 0, stream>>>(rp, pexcl, wp, N, E);
    k_scatter<<<(E + 255) / 256, 256, 0, stream>>>(row, col, ew, wp, ed, E);

    // MLP (WMMA bf16, LDS-staged B for GEMM1)
    dim3 g1((MT + 7) / 8, 2, 1);
    k_gemm1<<<g1, 256, 128 * BROW * sizeof(unsigned short), stream>>>(xb, w1t, b1, h1, MT);
    k_gemm2<<<MT / 2, 64, 0, stream>>>(h1, w2t, b2, hA);

    // K-step PPR propagation, ping-pong; last iteration writes d_out
    const float* hin = hA;
    for (int i = 0; i < KITER; ++i) {
        float* ho = (i == KITER - 1) ? out : ((i & 1) ? hA : hB);
        k_prop<<<N / 8, 256, 0, stream>>>(hin, ho, rp, ed);
        hin = ho;
    }
}
